// SelfAttention_44942537785930
// MI455X (gfx1250) — matrix operand order repaired and checked by reading, not compiled, not run on hardware
//
#include <hip/hip_runtime.h>

#define EMBED 1024
#define HEADS 16
#define HDIM  64
#define NB    2
#define SEQ   2048

typedef __attribute__((ext_vector_type(16))) _Float16 v16h;
typedef __attribute__((ext_vector_type(8)))  _Float16 v8h;
typedef __attribute__((ext_vector_type(8)))  float    v8f;

static __device__ __forceinline__ v16h cat8(v8h lo, v8h hi) {
  return __builtin_shufflevector(lo, hi, 0,1,2,3,4,5,6,7,8,9,10,11,12,13,14,15);
}

// ---------------------------------------------------------------------------
// Per-head projection: out[n,h,s,e] = sum_d x[n,s,h*64+d] * W[e,d], f16 output
// block = 256 threads = 4 seq rows x 64 output features
// ---------------------------------------------------------------------------
__global__ __launch_bounds__(256) void proj_kernel(const float* __restrict__ x,
                                                   const float* __restrict__ W,
                                                   _Float16* __restrict__ out) {
  __shared__ float xs[4][64];
  const int tid = threadIdx.x;
  const int e  = tid & 63;
  const int rl = tid >> 6;
  const int blk = blockIdx.x;               // N*H*(S/4)
  const int sb = blk % (SEQ / 4);
  const int nh = blk / (SEQ / 4);
  const int h  = nh % HEADS;
  const int n  = nh / HEADS;
  const int s  = sb * 4 + rl;

  xs[rl][e] = x[((long)(n * SEQ + s)) * EMBED + h * HDIM + e];
  __syncthreads();

  float acc = 0.f;
#pragma unroll
  for (int d = 0; d < HDIM; ++d)
    acc = fmaf(xs[rl][d], W[e * HDIM + d], acc);

  out[((long)nh * SEQ + s) * HDIM + e] = (_Float16)acc;
}

// ---------------------------------------------------------------------------
// f32 -> f16 elementwise convert (for Wo)
// ---------------------------------------------------------------------------
__global__ __launch_bounds__(256) void cvt_kernel(const float* __restrict__ src,
                                                  _Float16* __restrict__ dst, int nelem) {
  int i = blockIdx.x * 256 + threadIdx.x;
  if (i < nelem) dst[i] = (_Float16)src[i];
}

// ---------------------------------------------------------------------------
// Fused flash-attention over one (n,h,64-query) block.
// 128 threads = 4 waves; wave w owns query rows [qbase, qbase+16).
//  - next kv-block double-buffered through VGPRs (HBM latency hidden by
//    the compute phase; loads committed to LDS after the barrier)
//  - mask tile bit-packed wave-locally into a 512B LDS bitmask, so the
//    WMMA phase runs with low register pressure and batched fragments
// ---------------------------------------------------------------------------
__global__ __launch_bounds__(128) void attn_kernel(const _Float16* __restrict__ Qp,
                                                   const _Float16* __restrict__ Kp,
                                                   const _Float16* __restrict__ Vp,
                                                   const int* __restrict__ mask,
                                                   _Float16* __restrict__ Xo) {
  __shared__ _Float16 sK[64][72];     // [key][d]    natural
  __shared__ _Float16 sVt[64][72];    // [d][key]    transposed
  __shared__ _Float16 sP[4][16][72];  // per-wave P relayout buffer
  __shared__ unsigned sMask[64][2];   // bit k of [row][half] = (mask==0) at col half*32+k

  const int tid  = threadIdx.x;
  const int lane = tid & 31;
  const int w    = tid >> 5;
  const int lo16 = lane & 15;
  const int hi   = lane >> 4;   // 0 | 1

  const int blk  = blockIdx.x;
  const int qblk = blk & 31;    // S/64
  const int nh   = blk >> 5;
  const int h    = nh & (HEADS - 1);
  const int n    = nh >> 4;

  const long headBase = (long)nh * SEQ * HDIM;
  const int  qbase    = qblk * 64 + w * 16;

  // Q A-fragments: lane M = lo16; K offsets split by lane half (hi)
  v16h aq[2];
  {
    const _Float16* qrow = Qp + headBase + (long)(qbase + lo16) * HDIM;
#pragma unroll
    for (int c = 0; c < 2; ++c) {
      v8h l0 = *(const v8h*)(qrow + c * 32 + hi * 8);
      v8h l1 = *(const v8h*)(qrow + c * 32 + 16 + hi * 8);
      aq[c] = cat8(l0, l1);
    }
  }

  // staging registers for the next kv block (4 b128 chunks per thread)
  v8h kreg[4], vreg[4];
  int crow[4], ccol[4];
#pragma unroll
  for (int i = 0; i < 4; ++i) {
    int c = tid + 128 * i;
    crow[i] = c >> 3;
    ccol[i] = (c & 7) * 8;
  }

  // preload + stage block 0
#pragma unroll
  for (int i = 0; i < 4; ++i) {
    kreg[i] = *(const v8h*)(Kp + headBase + (long)crow[i] * HDIM + ccol[i]);
    vreg[i] = *(const v8h*)(Vp + headBase + (long)crow[i] * HDIM + ccol[i]);
  }
#pragma unroll
  for (int i = 0; i < 4; ++i) {
    *(v8h*)&sK[crow[i]][ccol[i]] = kreg[i];
#pragma unroll
    for (int e = 0; e < 8; ++e) sVt[ccol[i] + e][crow[i]] = vreg[i][e];
  }
  __syncthreads();

  // mask pack addressing: wave-local, lane packs half-row (lane>>1, lane&1)
  const int prow  = qblk * 64 + w * 16 + (lane >> 1);   // global q row
  const int prowl = w * 16 + (lane >> 1);               // local row in tile
  const int phalf = lane & 1;
  const int* mrowp = mask + ((long)(n * SEQ + prow)) * SEQ + phalf * 32;

  v8f o[4] = {};
  float m_[8], l_[8];
#pragma unroll
  for (int r = 0; r < 8; ++r) { m_[r] = -3.0e38f; l_[r] = 0.f; }

  for (int j = 0; j < SEQ / 64; ++j) {
    // (1) issue mask tile loads FIRST (they are waited on soon; LOADcnt
    //     decrements in issue order, so waiting for these does not drain
    //     the K/V loads issued after them)
    uint4 mw[8];
    {
      const int* mp = mrowp + j * 64;
#pragma unroll
      for (int q8 = 0; q8 < 8; ++q8)
        mw[q8] = *(const uint4*)(mp + q8 * 4);
    }

    // (2) issue next block's K/V global loads; in flight until the commit
    if (j + 1 < SEQ / 64) {
      const long nb = headBase + (long)((j + 1) * 64) * HDIM;
#pragma unroll
      for (int i = 0; i < 4; ++i) {
        kreg[i] = *(const v8h*)(Kp + nb + (long)crow[i] * HDIM + ccol[i]);
        vreg[i] = *(const v8h*)(Vp + nb + (long)crow[i] * HDIM + ccol[i]);
      }
    }

    // (3) pack mask==0 predicates into one dword, store to wave-local LDS
    {
      unsigned bits = 0;
#pragma unroll
      for (int q8 = 0; q8 < 8; ++q8) {
        bits |= (mw[q8].x == 0u ? 1u : 0u) << (q8 * 4 + 0);
        bits |= (mw[q8].y == 0u ? 1u : 0u) << (q8 * 4 + 1);
        bits |= (mw[q8].z == 0u ? 1u : 0u) << (q8 * 4 + 2);
        bits |= (mw[q8].w == 0u ? 1u : 0u) << (q8 * 4 + 3);
      }
      sMask[prowl][phalf] = bits;
    }

    // (4) S = Q K^T : per contraction chunk, 4 B fragments then 4 WMMAs
    v8f ct[4] = {};
#pragma unroll
    for (int c = 0; c < 2; ++c) {
      v16h bk[4];
#pragma unroll
      for (int t = 0; t < 4; ++t) {
        const _Float16* kr = &sK[t * 16 + lo16][c * 32 + hi * 8];
        bk[t] = cat8(*(const v8h*)kr, *(const v8h*)(kr + 16));
      }
#pragma unroll
      for (int t = 0; t < 4; ++t)
        ct[t] = __builtin_amdgcn_wmma_f32_16x16x32_f16(false, aq[c], false, bk[t],
                                                       (short)0, ct[t], false, false);
    }

    // (5) masked fill (faithful: +1e20 where mask==0), scale by 1/32,
    //     online softmax with 16-lane-half reductions
    unsigned long long mm[8];
#pragma unroll
    for (int r = 0; r < 8; ++r)
      mm[r] = *(const unsigned long long*)&sMask[w * 16 + r + hi * 8][0];

    float p[4][8];
#pragma unroll
    for (int t = 0; t < 4; ++t) {
      const int bitpos = (t & 1) * 16 + lo16;
#pragma unroll
      for (int r = 0; r < 8; ++r) {
        unsigned word = (t < 2) ? (unsigned)mm[r] : (unsigned)(mm[r] >> 32);
        float e = ct[t][r];
        e = ((word >> bitpos) & 1u) ? 1.0e20f : e;
        p[t][r] = e * 0.03125f;
      }
    }
#pragma unroll
    for (int r = 0; r < 8; ++r) {
      float loc = fmaxf(fmaxf(p[0][r], p[1][r]), fmaxf(p[2][r], p[3][r]));
#pragma unroll
      for (int off = 1; off < 16; off <<= 1)
        loc = fmaxf(loc, __shfl_xor(loc, off, 32));
      float nm = fmaxf(m_[r], loc);
      float sc = __expf(m_[r] - nm);
      float rs = 0.f;
#pragma unroll
      for (int t = 0; t < 4; ++t) {
        float pv = __expf(p[t][r] - nm);
        p[t][r] = pv;
        rs += pv;
      }
#pragma unroll
      for (int off = 1; off < 16; off <<= 1)
        rs += __shfl_xor(rs, off, 32);
      l_[r] = l_[r] * sc + rs;
      m_[r] = nm;
#pragma unroll
      for (int t2 = 0; t2 < 4; ++t2) o[t2][r] *= sc;
    }

    // (6) relayout P (C-layout -> A-layout) through per-wave LDS tile
#pragma unroll
    for (int t = 0; t < 4; ++t)
#pragma unroll
      for (int r = 0; r < 8; ++r)
        sP[w][r + hi * 8][t * 16 + lo16] = (_Float16)p[t][r];

    // (7) O += P V : per contraction chunk, A + 4 B fragments, 4 WMMAs
    //     (same-wave LDS ops are in-order; no barrier for the sP roundtrip)
#pragma unroll
    for (int c = 0; c < 2; ++c) {
      const _Float16* pr = &sP[w][lo16][c * 32 + hi * 8];
      v16h ap = cat8(*(const v8h*)pr, *(const v8h*)(pr + 16));
      v16h bv[4];
#pragma unroll
      for (int t2 = 0; t2 < 4; ++t2) {
        const _Float16* vr = &sVt[t2 * 16 + lo16][c * 32 + hi * 8];
        bv[t2] = cat8(*(const v8h*)vr, *(const v8h*)(vr + 16));
      }
#pragma unroll
      for (int t2 = 0; t2 < 4; ++t2)
        o[t2] = __builtin_amdgcn_wmma_f32_16x16x32_f16(false, ap, false, bv[t2],
                                                       (short)0, o[t2], false, false);
    }
    __syncthreads();

    // (8) commit staged next block into LDS (first point that waits on the
    //     K/V loads — their latency was hidden by the whole compute phase)
    if (j + 1 < SEQ / 64) {
#pragma unroll
      for (int i = 0; i < 4; ++i) {
        *(v8h*)&sK[crow[i]][ccol[i]] = kreg[i];
#pragma unroll
        for (int e = 0; e < 8; ++e) sVt[ccol[i] + e][crow[i]] = vreg[i][e];
      }
    }
    __syncthreads();
  }

  // normalize and store to [n, q, E] f16 staging for output GEMM
  float inv[8];
#pragma unroll
  for (int r = 0; r < 8; ++r) inv[r] = 1.0f / l_[r];
#pragma unroll
  for (int t2 = 0; t2 < 4; ++t2) {
#pragma unroll
    for (int r = 0; r < 8; ++r) {
      int qrow = qbase + r + hi * 8;
      int d    = t2 * 16 + lo16;
      Xo[((long)(n * SEQ + qrow)) * EMBED + h * HDIM + d] = (_Float16)(o[t2][r] * inv[r]);
    }
  }
}

// ---------------------------------------------------------------------------
// Output projection: out[row,col] = sum_k X[row,k]*Wo[col,k] + bo[col]
// Wo kept natural [e_out][e_in]: B-fragment reads are contiguous row segments.
// ---------------------------------------------------------------------------
__global__ __launch_bounds__(128) void ogemm_kernel(const _Float16* __restrict__ X,
                                                    const _Float16* __restrict__ Wh,
                                                    const float* __restrict__ bo,
                                                    float* __restrict__ out) {
  const int tid  = threadIdx.x;
  const int lane = tid & 31;
  const int w    = tid >> 5;
  const int lo16 = lane & 15;
  const int hi   = lane >> 4;
  const int bc   = blockIdx.x & 15;   // EMBED/64 col blocks
  const int br   = blockIdx.x >> 4;   // (N*SEQ)/64 row blocks
  const int rbase = br * 64 + w * 16;
  const int cbase = bc * 64;

  v8f acc[4] = {};
  const _Float16* xrow = X + (long)(rbase + lo16) * EMBED;
  for (int kc = 0; kc < EMBED / 32; ++kc) {
    v16h a = cat8(*(const v8h*)(xrow + kc * 32 + hi * 8),
                  *(const v8h*)(xrow + kc * 32 + 16 + hi * 8));
    v16h b[4];
#pragma unroll
    for (int t = 0; t < 4; ++t) {
      const _Float16* wr = Wh + (long)(cbase + t * 16 + lo16) * EMBED + kc * 32 + hi * 8;
      b[t] = cat8(*(const v8h*)wr, *(const v8h*)(wr + 16));
    }
#pragma unroll
    for (int t = 0; t < 4; ++t)
      acc[t] = __builtin_amdgcn_wmma_f32_16x16x32_f16(false, a, false, b[t],
                                                      (short)0, acc[t], false, false);
  }
#pragma unroll
  for (int t = 0; t < 4; ++t) {
    int col = cbase + t * 16 + lo16;
    float bias = bo[col];
#pragma unroll
    for (int r = 0; r < 8; ++r) {
      int row = rbase + r + hi * 8;
      out[(long)row * EMBED + col] = acc[t][r] + bias;
    }
  }
}

// ---------------------------------------------------------------------------
extern "C" void kernel_launch(void* const* d_in, const int* in_sizes, int n_in,
                              void* d_out, int out_size, void* d_ws, size_t ws_size,
                              hipStream_t stream) {
  const float* value = (const float*)d_in[0];
  const float* key   = (const float*)d_in[1];
  const float* query = (const float*)d_in[2];
  const int*   mask  = (const int*)d_in[3];
  const float* Wv    = (const float*)d_in[4];
  const float* Wk    = (const float*)d_in[5];
  const float* Wq    = (const float*)d_in[6];
  const float* Wo    = (const float*)d_in[7];
  const float* bo    = (const float*)d_in[8];
  float* out = (float*)d_out;

  char* ws = (char*)d_ws;
  const size_t PROJ = (size_t)NB * HEADS * SEQ * HDIM;  // 4,194,304 elems
  _Float16* Qp = (_Float16*)ws;  ws += PROJ * 2;
  _Float16* Kp = (_Float16*)ws;  ws += PROJ * 2;
  _Float16* Vp = (_Float16*)ws;  ws += PROJ * 2;
  _Float16* Xo = (_Float16*)ws;  ws += (size_t)NB * SEQ * EMBED * 2;
  _Float16* Wh = (_Float16*)ws;  ws += (size_t)EMBED * EMBED * 2;

  const int projBlocks = NB * HEADS * (SEQ / 4);  // 16384
  proj_kernel<<<projBlocks, 256, 0, stream>>>(query, Wq, Qp);
  proj_kernel<<<projBlocks, 256, 0, stream>>>(key,   Wk, Kp);
  proj_kernel<<<projBlocks, 256, 0, stream>>>(value, Wv, Vp);
  cvt_kernel<<<(EMBED * EMBED + 255) / 256, 256, 0, stream>>>(Wo, Wh, EMBED * EMBED);

  attn_kernel<<<NB * HEADS * (SEQ / 64), 128, 0, stream>>>(Qp, Kp, Vp, mask, Xo);
  ogemm_kernel<<<(NB * SEQ / 64) * (EMBED / 64), 128, 0, stream>>>(Xo, Wh, bo, out);
}
